// DistillCodecEncoder_32796370272320
// MI455X (gfx1250) — matrix operand order repaired and checked by reading, not compiled, not run on hardware
//
#include <hip/hip_runtime.h>
#include <cstdint>

typedef __bf16 bf16_t;
typedef __attribute__((ext_vector_type(16))) __bf16 v16bf;
typedef __attribute__((ext_vector_type(8)))  float  v8f;
typedef int v4i __attribute__((vector_size(16)));

#define DIMC   128
#define NHEADS 6
#define DHEAD  32
#define INNER  192
#define FF2    682
#define FF2P   688
#define FFI    341
#define FFP    352
#define TIN    12800
#define NBATCH 2
#define WIN1   406
#define NWIN   32
#define NSEQ1  (WIN1*NWIN)   /* 12992 */
#define WIN2   203
#define NSEQ2  (WIN2*NWIN)   /* 6496 */

union AF { v16bf v; uint4 q[2]; bf16_t e[16]; };

static __device__ __forceinline__ v8f wmma_bf16(v16bf a, v16bf b, v8f c) {
  return __builtin_amdgcn_wmma_f32_16x16x32_bf16(false, a, false, b, (short)0, c, false, false);
}

#ifndef __has_builtin
#define __has_builtin(x) 0
#endif

// async global->LDS copy of 16B per lane (CDNA5), with inline-asm fallback
#if __has_builtin(__builtin_amdgcn_global_load_async_to_lds_b128)
#define ASYNC_CP16(gp, lp)                                                     \
  __builtin_amdgcn_global_load_async_to_lds_b128(                              \
      (__attribute__((address_space(1))) v4i*)(gp),                            \
      (__attribute__((address_space(3))) v4i*)(lp), 0, 0)
#else
#define ASYNC_CP16(gp, lp)                                                     \
  asm volatile("global_load_async_to_lds_b128 %0, %1, off" ::                  \
               "v"((uint32_t)(uintptr_t)(__attribute__((address_space(3))) char*)(lp)), \
               "v"((const void*)(gp)) : "memory")
#endif

#if __has_builtin(__builtin_amdgcn_s_wait_asynccnt)
#define WAIT_ASYNC(n) __builtin_amdgcn_s_wait_asynccnt(n)
#else
#define WAIT_ASYNC(n) asm volatile("s_wait_asynccnt %0" :: "i"(n) : "memory")
#endif

// ---------------------------------------------------------------- rearrange
__global__ void rearrange_kernel(const float* __restrict__ x,
                                 const float* __restrict__ cache,
                                 float* __restrict__ out, int total) {
  int idx = blockIdx.x * blockDim.x + threadIdx.x;
  if (idx >= total) return;
  int c = idx & (DIMC - 1);
  int j = (idx / DIMC) % NSEQ1;
  int b = idx / (DIMC * NSEQ1);
  int chunk = j / WIN1;
  int r = j - chunk * WIN1;
  float v;
  if (r < 6) v = cache[r * DIMC + c];
  else       v = x[((size_t)b * DIMC + c) * TIN + (chunk * 400 + (r - 6))];
  out[idx] = v;
}

// ------------------------------------------------- weight convert (K,N)->bf16 (N,Kpad)
__global__ void convert_wT_kernel(const float* __restrict__ W, bf16_t* __restrict__ Bt,
                                  int K, int Nout, int Kpad, int total) {
  int idx = blockIdx.x * blockDim.x + threadIdx.x;
  if (idx >= total) return;
  int n = idx / Kpad;
  int k = idx - n * Kpad;
  float v = (k < K && n < Nout) ? W[(size_t)k * Nout + n] : 0.f;
  Bt[idx] = (bf16_t)v;
}

__global__ void convert_conv_kernel(const float* __restrict__ W, bf16_t* __restrict__ Bt) {
  int idx = blockIdx.x * blockDim.x + threadIdx.x;
  if (idx >= DIMC * 256) return;
  int co = idx >> 8;
  int k  = idx & 255;
  int ci = k & 127;
  int kk = k >> 7;
  Bt[idx] = (bf16_t)W[((size_t)co * DIMC + ci) * 2 + kk];
}

// ---------------------------------------------------------------- layernorm -> bf16
__global__ __launch_bounds__(32)
void ln_kernel(const float* __restrict__ x, const float* __restrict__ g,
               const float* __restrict__ bta, bf16_t* __restrict__ y) {
  int row = blockIdx.x;
  int lane = threadIdx.x;
  const float* xr = x + (size_t)row * DIMC;
  float4 v = *(const float4*)(xr + lane * 4);
  float s = v.x + v.y + v.z + v.w;
  for (int o = 16; o >= 1; o >>= 1) s += __shfl_xor(s, o, 32);
  float mean = s * (1.f / DIMC);
  float d0 = v.x - mean, d1 = v.y - mean, d2 = v.z - mean, d3 = v.w - mean;
  float qq = d0 * d0 + d1 * d1 + d2 * d2 + d3 * d3;
  for (int o = 16; o >= 1; o >>= 1) qq += __shfl_xor(qq, o, 32);
  float rstd = rsqrtf(qq * (1.f / DIMC) + 1e-5f);
  bf16_t* yr = y + (size_t)row * DIMC;
  int i = lane * 4;
  yr[i + 0] = (bf16_t)(d0 * rstd * g[i + 0] + bta[i + 0]);
  yr[i + 1] = (bf16_t)(d1 * rstd * g[i + 1] + bta[i + 1]);
  yr[i + 2] = (bf16_t)(d2 * rstd * g[i + 2] + bta[i + 2]);
  yr[i + 3] = (bf16_t)(d3 * rstd * g[i + 3] + bta[i + 3]);
}

// ---------------------------------------------------------------- WMMA GEMM
// A: (M,K) bf16 row-major lda, Bt: (N,Kpad) bf16 row-major ldb(=Kpad)
// Block = 4 waves stacked along M (128 rows); shared NT*16-wide B strip staged
// in LDS via async copies (double buffered). Each wave computes a 32 x NT*16
// register tile. NT is compile-time so the hot loop is branch-free.
// MODE 0: C = A*B ; MODE 1: C += A*B ; MODE 2: C = A*B + bias[n]
template<int MODE, int NT>
__global__ __launch_bounds__(128)
void gemm_bf16(const bf16_t* __restrict__ A, int lda,
               const bf16_t* __restrict__ Bt, int ldb,
               float* __restrict__ C, int ldc,
               const float* __restrict__ bias,
               int M, int N, int K, int nblk0) {
  __shared__ bf16_t btile[2][64][32];
  int tid  = threadIdx.x;
  int wave = tid >> 5;
  int lane = tid & 31;
  int half = lane >> 4;
  int l16  = lane & 15;
  int nblk = nblk0 + blockIdx.x * 64;
  int mblk = blockIdx.y * 128;
  int mw   = mblk + wave * 32;

  int r0 = mw + l16;      if (r0 > M - 1) r0 = M - 1;
  int r1 = mw + 16 + l16; if (r1 > M - 1) r1 = M - 1;
  const bf16_t* arow0 = A + (size_t)r0 * lda;
  const bf16_t* arow1 = A + (size_t)r1 * lda;

  v8f zero = {};
  v8f acc[2][NT];
#pragma unroll
  for (int i = 0; i < 2; ++i)
#pragma unroll
    for (int j = 0; j < NT; ++j) acc[i][j] = zero;

  // stage one 64x32 bf16 B tile (4KB) into LDS buffer `buf` for k-offset `k`
  auto stage = [&](int buf, int k) {
#pragma unroll
    for (int it = 0; it < 2; ++it) {
      int idx = it * 128 + tid;        // 0..255 -> 256 x 16B = 4KB
      int row = idx >> 2;              // 0..63
      int seg = idx & 3;
      int gn = nblk + row; if (gn > N - 1) gn = N - 1;
      const bf16_t* gp = Bt + (size_t)gn * ldb + k + seg * 8;
      ASYNC_CP16(gp, &btile[buf][row][seg * 8]);
    }
  };

  stage(0, 0);
  for (int k = 0; k < K; k += 32) {
    int kn = k + 32;
    int cur = (k >> 5) & 1;
    if (kn < K) stage(cur ^ 1, kn);
    if (kn < K) WAIT_ASYNC(2);
    else        WAIT_ASYNC(0);
    __syncthreads();

    AF a0, a1;
    a0.q[0] = *(const uint4*)(arow0 + k + half * 8);
    a0.q[1] = *(const uint4*)(arow0 + k + 16 + half * 8);
    a1.q[0] = *(const uint4*)(arow1 + k + half * 8);
    a1.q[1] = *(const uint4*)(arow1 + k + 16 + half * 8);
    __builtin_prefetch(arow0 + k + 32, 0, 1);
    __builtin_prefetch(arow1 + k + 32, 0, 1);

#pragma unroll
    for (int nt = 0; nt < NT; ++nt) {
      AF b;
      const bf16_t* brow = &btile[cur][nt * 16 + l16][0];
      b.q[0] = *(const uint4*)(brow + half * 8);
      b.q[1] = *(const uint4*)(brow + 16 + half * 8);
      acc[0][nt] = wmma_bf16(a0.v, b.v, acc[0][nt]);
      acc[1][nt] = wmma_bf16(a1.v, b.v, acc[1][nt]);
    }
    __syncthreads();
  }

#pragma unroll
  for (int mt = 0; mt < 2; ++mt) {
    int mbase = mw + mt * 16;
    if (mbase < M) {                         // M is a multiple of 16
#pragma unroll
      for (int nt = 0; nt < NT; ++nt) {
        int n = nblk + nt * 16 + l16;
#pragma unroll
        for (int r = 0; r < 8; ++r) {
          int m = mbase + r + half * 8;
          size_t o = (size_t)m * ldc + n;
          float vv = acc[mt][nt][r];
          if (MODE == 1)      C[o] += vv;
          else if (MODE == 2) C[o] = vv + bias[n];
          else                C[o] = vv;
        }
      }
    }
  }
}

// ---------------------------------------------------- qkv split + rotary -> bf16 heads
__global__ void qkv_rope_kernel(const float* __restrict__ qkv,
                                bf16_t* __restrict__ qh, bf16_t* __restrict__ kh,
                                bf16_t* __restrict__ vh, int Nseq) {
  int g = blockIdx.x;
  int t = threadIdx.x;               // 0..191
  int h = t >> 5;
  int d = t & 31;
  int b = g / Nseq;
  int n = g - b * Nseq;
  const float* row = qkv + (size_t)g * (3 * INNER);
  int col = h * DHEAD + d;
  float qv = row[col];
  float kv = row[INNER + col];
  float vv = row[2 * INNER + col];
  int pcol = (d < 16) ? col + 16 : col - 16;
  float qp = row[pcol];
  float kp = row[INNER + pcol];
  float sgn = (d < 16) ? -1.f : 1.f;
  float inv = __powf(10000.f, -(float)(d & 15) * (1.f / 16.f));
  float ang = (float)n * inv;
  float sn, cs;
  __sincosf(ang, &sn, &cs);
  size_t o = (((size_t)b * NHEADS + h) * Nseq + n) * DHEAD + d;
  qh[o] = (bf16_t)(qv * cs + sgn * qp * sn);
  kh[o] = (bf16_t)(kv * cs + sgn * kp * sn);
  vh[o] = (bf16_t)vv;
}

// ---------------------------------------------------------------- local attention
// one wave per (batch*head, window, 16-query tile); KMAX = padded key count
template<int KMAX>
__global__ __launch_bounds__(32)
void attn_kernel(const bf16_t* __restrict__ qh, const bf16_t* __restrict__ kh,
                 const bf16_t* __restrict__ vh, bf16_t* __restrict__ attO,
                 int Nseq, int win) {
  __shared__ float sim[16][KMAX];
  __shared__ float rsum[16];
  int bh = blockIdx.x;
  int b = bh / NHEADS;
  int h = bh - b * NHEADS;
  int wi = blockIdx.y;
  int m0 = blockIdx.z * 16;
  int lane = threadIdx.x;
  int half = lane >> 4;
  int l16 = lane & 15;
  int ws = wi * win;
  int nk = 2 * win;
  int nkc = (nk + 31) >> 5;
  const bf16_t* qb = qh + (size_t)bh * Nseq * DHEAD;
  const bf16_t* kb = kh + (size_t)bh * Nseq * DHEAD;
  const bf16_t* vb = vh + (size_t)bh * Nseq * DHEAD;

  // q A-fragment (rows = queries, K = head dim 32)
  int qrow = ws + m0 + l16;
  int qcl = qrow < Nseq ? qrow : Nseq - 1;
  AF afr;
  afr.q[0] = *(const uint4*)(qb + (size_t)qcl * DHEAD + half * 8);
  afr.q[1] = *(const uint4*)(qb + (size_t)qcl * DHEAD + 16 + half * 8);

  const float scale = 0.17677669529663687f;   // 1/sqrt(32)
  for (int kc = 0; kc < nkc; ++kc) {
#pragma unroll
    for (int nt = 0; nt < 2; ++nt) {
      int jb = kc * 32 + nt * 16;
      int kpos = ws - win + jb + l16;
      int kcl = kpos;
      if (kcl < 0) kcl = 0;
      if (kcl >= Nseq) kcl = Nseq - 1;
      AF bfr;
      bfr.q[0] = *(const uint4*)(kb + (size_t)kcl * DHEAD + half * 8);
      bfr.q[1] = *(const uint4*)(kb + (size_t)kcl * DHEAD + 16 + half * 8);
      v8f c = {};
      c = wmma_bf16(afr.v, bfr.v, c);
#pragma unroll
      for (int r = 0; r < 8; ++r) {
        int m = r + half * 8;
        int tq = ws + m0 + m;
        int j = jb + l16;
        int kp = ws - win + j;
        bool ok = (j < nk) && (kp >= 0) && (kp <= tq) && (m0 + m < win);
        sim[m][j] = ok ? c[r] * scale : -3.0e38f;
      }
    }
  }
  __syncthreads();
  if (lane < 16) {
    int row = lane;
    int nn = nkc * 32;
    if (m0 + row >= win) {
      for (int j = 0; j < nn; ++j) sim[row][j] = 0.f;
      rsum[row] = 1.f;
    } else {
      float mx = -3.0e38f;
      for (int j = 0; j < nn; ++j) mx = fmaxf(mx, sim[row][j]);
      float sm = 0.f;
      for (int j = 0; j < nn; ++j) {
        float e = __expf(sim[row][j] - mx);
        sim[row][j] = e;
        sm += e;
      }
      rsum[row] = sm;
    }
  }
  __syncthreads();

  v8f o0 = {}, o1 = {};
  for (int kc = 0; kc < nkc; ++kc) {
    AF p;
    const float* srow = &sim[l16][kc * 32];
#pragma unroll
    for (int j = 0; j < 16; ++j) {
      int off = half * 16 + j;
      p.e[j] = (bf16_t)srow[off];
    }
    int ks0 = ws - win + kc * 32 + half * 16;
#pragma unroll
    for (int nt = 0; nt < 2; ++nt) {
      int dh = nt * 16 + l16;
      AF vf;
#pragma unroll
      for (int jj = 0; jj < 16; ++jj) {
        int kp = ks0 + jj;
        if (kp < 0) kp = 0;
        if (kp >= Nseq) kp = Nseq - 1;
        vf.e[jj] = vb[(size_t)kp * DHEAD + dh];
      }
      if (nt == 0) o0 = wmma_bf16(p.v, vf.v, o0);
      else         o1 = wmma_bf16(p.v, vf.v, o1);
    }
  }
#pragma unroll
  for (int r = 0; r < 8; ++r) {
    int m = r + half * 8;
    int qi = m0 + m;
    if (qi < win) {
      float inv = 1.f / rsum[m];
      size_t qpos = (size_t)ws + qi;
      size_t base = ((size_t)b * Nseq + qpos) * INNER + h * DHEAD;
      attO[base + l16]      = (bf16_t)(o0[r] * inv);
      attO[base + 16 + l16] = (bf16_t)(o1[r] * inv);
    }
  }
}

// ---------------------------------------------------------------- GEGLU
__global__ void geglu_kernel(const float* __restrict__ hbuf, bf16_t* __restrict__ g) {
  int row = blockIdx.x;
  int j = threadIdx.x;                  // 0..351
  float v = 0.f;
  if (j < FFI) {
    float a  = hbuf[(size_t)row * FF2P + j];
    float gt = hbuf[(size_t)row * FF2P + FFI + j];
    v = a * 0.5f * gt * (1.f + erff(gt * 0.70710678118654752f));
  }
  g[(size_t)row * FFP + j] = (bf16_t)v;
}

__global__ void cvt_bf16_kernel(const float* __restrict__ x, bf16_t* __restrict__ y, int total) {
  int i = blockIdx.x * blockDim.x + threadIdx.x;
  if (i < total) y[i] = (bf16_t)x[i];
}

// ================================================================ launch
extern "C" void kernel_launch(void* const* d_in, const int* in_sizes, int n_in,
                              void* d_out, int out_size, void* d_ws, size_t ws_size,
                              hipStream_t stream) {
  (void)in_sizes; (void)n_in; (void)out_size; (void)ws_size;
  const float* x        = (const float*)d_in[0];
  const float* cache    = (const float*)d_in[1];
  const float* dt_ln1_w = (const float*)d_in[2];
  const float* dt_ln1_b = (const float*)d_in[3];
  const float* dt_wqkv  = (const float*)d_in[4];
  const float* dt_wo    = (const float*)d_in[5];
  const float* dt_ln2_w = (const float*)d_in[6];
  const float* dt_ln2_b = (const float*)d_in[7];
  const float* dt_w1    = (const float*)d_in[8];
  const float* dt_w2    = (const float*)d_in[9];
  const float* lt_ln1_w = (const float*)d_in[10];
  const float* lt_ln1_b = (const float*)d_in[11];
  const float* lt_wqkv  = (const float*)d_in[12];
  const float* lt_wo    = (const float*)d_in[13];
  const float* lt_ln2_w = (const float*)d_in[14];
  const float* lt_ln2_b = (const float*)d_in[15];
  const float* lt_w1    = (const float*)d_in[16];
  const float* lt_w2    = (const float*)d_in[17];
  const float* conv_w   = (const float*)d_in[18];
  const float* conv_b   = (const float*)d_in[19];
  float* out = (float*)d_out;

  char* ws = (char*)d_ws;
  size_t off = 0;
  auto alloc = [&](size_t bytes) -> void* {
    off = (off + 255) & ~(size_t)255;
    void* p = ws + off;
    off += bytes;
    return p;
  };
  const size_t Mtok1 = (size_t)NBATCH * NSEQ1;
  float*  bufA = (float*) alloc(Mtok1 * DIMC * 4);
  bf16_t* xln  = (bf16_t*)alloc(Mtok1 * DIMC * 2);
  float*  bufH = (float*) alloc(Mtok1 * FF2P * 4);
  bf16_t* qh   = (bf16_t*)alloc(Mtok1 * INNER * 2);
  bf16_t* kh   = (bf16_t*)alloc(Mtok1 * INNER * 2);
  bf16_t* vh   = (bf16_t*)alloc(Mtok1 * INNER * 2);
  bf16_t* attO = (bf16_t*)alloc(Mtok1 * INNER * 2);
  bf16_t* gbuf = qh;   // alias: q/k head bufs dead by GEGLU time; 352 <= 2*192 elems/row
  bf16_t* WqkvT[4]; bf16_t* WoT[4]; bf16_t* W1T[4]; bf16_t* W2T[4];
  for (int i = 0; i < 4; ++i) {
    WqkvT[i] = (bf16_t*)alloc((size_t)576 * 128 * 2);
    WoT[i]   = (bf16_t*)alloc((size_t)128 * 192 * 2);
    W1T[i]   = (bf16_t*)alloc((size_t)688 * 128 * 2);
    W2T[i]   = (bf16_t*)alloc((size_t)128 * 352 * 2);
  }
  bf16_t* convT = (bf16_t*)alloc((size_t)128 * 256 * 2);

  // weight conversion (f32 -> bf16, transposed to N x Kpad)
  const float* Wq[4]  = { dt_wqkv, dt_wqkv + 128 * 576, lt_wqkv, lt_wqkv + 128 * 576 };
  const float* Wo_[4] = { dt_wo,   dt_wo   + 192 * 128, lt_wo,   lt_wo   + 192 * 128 };
  const float* W1_[4] = { dt_w1,   dt_w1   + 128 * 682, lt_w1,   lt_w1   + 128 * 682 };
  const float* W2_[4] = { dt_w2,   dt_w2   + 341 * 128, lt_w2,   lt_w2   + 341 * 128 };
  for (int i = 0; i < 4; ++i) {
    { int tot = 576 * 128; convert_wT_kernel<<<(tot + 255) / 256, 256, 0, stream>>>(Wq[i],  WqkvT[i], 128, 576, 128, tot); }
    { int tot = 128 * 192; convert_wT_kernel<<<(tot + 255) / 256, 256, 0, stream>>>(Wo_[i], WoT[i],   192, 128, 192, tot); }
    { int tot = 688 * 128; convert_wT_kernel<<<(tot + 255) / 256, 256, 0, stream>>>(W1_[i], W1T[i],   128, 682, 128, tot); }
    { int tot = 128 * 352; convert_wT_kernel<<<(tot + 255) / 256, 256, 0, stream>>>(W2_[i], W2T[i],   341, 128, 352, tot); }
  }
  convert_conv_kernel<<<(128 * 256 + 255) / 256, 256, 0, stream>>>(conv_w, convT);

  // interleave cache tokens per 400-sample chunk
  { int tot = (int)(Mtok1 * DIMC);
    rearrange_kernel<<<(tot + 255) / 256, 256, 0, stream>>>(x, cache, bufA, tot); }

  auto run_layer = [&](float* act, int Nseq, int win, int qtiles, int kset, int li,
                       const float* l1w, const float* l1b,
                       const float* l2w, const float* l2b) {
    int Mt = NBATCH * Nseq;
    int gy = (Mt + 127) / 128;
    ln_kernel<<<Mt, 32, 0, stream>>>(act, l1w, l1b, xln);
    gemm_bf16<0, 4><<<dim3(9, gy), 128, 0, stream>>>(
        xln, DIMC, WqkvT[li], DIMC, bufH, 3 * INNER, nullptr, Mt, 3 * INNER, DIMC, 0);
    qkv_rope_kernel<<<Mt, INNER, 0, stream>>>(bufH, qh, kh, vh, Nseq);
    dim3 ag(NBATCH * NHEADS, NWIN, qtiles);
    if (kset == 0) attn_kernel<832><<<ag, 32, 0, stream>>>(qh, kh, vh, attO, Nseq, win);
    else           attn_kernel<416><<<ag, 32, 0, stream>>>(qh, kh, vh, attO, Nseq, win);
    gemm_bf16<1, 4><<<dim3(2, gy), 128, 0, stream>>>(
        attO, INNER, WoT[li], INNER, act, DIMC, nullptr, Mt, DIMC, INNER, 0);
    ln_kernel<<<Mt, 32, 0, stream>>>(act, l2w, l2b, xln);
    // w1: N=688 = 10*64 (full, NT=4) + 48 (tail, NT=3)
    gemm_bf16<0, 4><<<dim3(10, gy), 128, 0, stream>>>(
        xln, DIMC, W1T[li], DIMC, bufH, FF2P, nullptr, Mt, FF2P, DIMC, 0);
    gemm_bf16<0, 3><<<dim3(1, gy), 128, 0, stream>>>(
        xln, DIMC, W1T[li], DIMC, bufH, FF2P, nullptr, Mt, FF2P, DIMC, 640);
    geglu_kernel<<<Mt, FFP, 0, stream>>>(bufH, gbuf);
    gemm_bf16<1, 4><<<dim3(2, gy), 128, 0, stream>>>(
        gbuf, FFP, W2T[li], FFP, act, DIMC, nullptr, Mt, DIMC, FFP, 0);
  };

  // DownTrans: 2 layers, window 406
  run_layer(bufA, NSEQ1, WIN1, 26, 0, 0, dt_ln1_w,        dt_ln1_b,        dt_ln2_w,        dt_ln2_b);
  run_layer(bufA, NSEQ1, WIN1, 26, 0, 1, dt_ln1_w + DIMC, dt_ln1_b + DIMC, dt_ln2_w + DIMC, dt_ln2_b + DIMC);

  // strided conv compress (kernel 2 / stride 2) as GEMM over paired rows
  { int tot = (int)(Mtok1 * DIMC);
    cvt_bf16_kernel<<<(tot + 255) / 256, 256, 0, stream>>>(bufA, xln, tot); }
  for (int b = 0; b < NBATCH; ++b) {
    gemm_bf16<2, 4><<<dim3(2, (NSEQ2 + 127) / 128), 128, 0, stream>>>(
        xln + (size_t)b * NSEQ1 * DIMC, 2 * DIMC, convT, 2 * DIMC,
        out + (size_t)b * NSEQ2 * DIMC, DIMC, conv_b, NSEQ2, DIMC, 2 * DIMC, 0);
  }

  // LocalTrans: 2 layers, window 203 — in place on d_out
  run_layer(out, NSEQ2, WIN2, 13, 1, 2, lt_ln1_w,        lt_ln1_b,        lt_ln2_w,        lt_ln2_b);
  run_layer(out, NSEQ2, WIN2, 13, 1, 3, lt_ln1_w + DIMC, lt_ln1_b + DIMC, lt_ln2_w + DIMC, lt_ln2_b + DIMC);
}
